// BiMambaEncoder_40939628266181
// MI455X (gfx1250) — matrix operand order repaired and checked by reading, not compiled, not run on hardware
//
#include <hip/hip_runtime.h>
#include <hip/hip_bf16.h>
#include <math.h>
#include <stdint.h>

// ---------------- model constants ----------------
#define B_      2
#define L_      1024
#define DMODEL  256
#define DSTATE  64
#define DCONV   4
#define DINNER  512            // EXPAND * DMODEL
#define NHEADS  8              // DINNER / HEADDIM
#define HEADDIM 64
#define CONVDIM 640            // DINNER + 2*DSTATE
#define DINPROJ 1160           // 2*DINNER + 2*DSTATE + NHEADS
#define EPS_    1e-5f
#define BLrows  (B_ * L_)      // 2048
#define KMAX    512            // largest GEMM K (out_proj / fusion)

typedef __attribute__((ext_vector_type(16))) _Float16 v16h;
typedef __attribute__((ext_vector_type(8)))  _Float16 v8h;
typedef __attribute__((ext_vector_type(8)))  float    v8f;
typedef __attribute__((ext_vector_type(4)))  unsigned int u32x4;
typedef __attribute__((ext_vector_type(8)))  int          i32x8;
typedef __attribute__((ext_vector_type(4)))  int          i32x4;

// ---- Tensor Data Mover availability / arity (ROCm7.2 clang-22: 5 args; clang-23: 6 args) ----
#if defined(__has_builtin)
#  if __has_builtin(__builtin_amdgcn_tensor_load_to_lds)
#    define HAS_TDM 1
#  endif
#endif
#ifndef HAS_TDM
#  define HAS_TDM 0
#endif
#if HAS_TDM
#  if defined(__clang_major__) && (__clang_major__ >= 23)
#    define TDM_LOAD(g0, g1, g2, g3) \
        __builtin_amdgcn_tensor_load_to_lds((g0), (g1), (g2), (g3), (i32x8){0,0,0,0,0,0,0,0}, 0)
#  else
#    define TDM_LOAD(g0, g1, g2, g3) \
        __builtin_amdgcn_tensor_load_to_lds((g0), (g1), (g2), (g3), 0)
#  endif
#endif
#if defined(__has_builtin)
#  if __has_builtin(__builtin_amdgcn_s_wait_tensorcnt)
#    define WAIT_TENSORCNT0() __builtin_amdgcn_s_wait_tensorcnt(0)
#  endif
#endif
#ifndef WAIT_TENSORCNT0
#  define WAIT_TENSORCNT0()
#endif

__device__ __forceinline__ float sigmoidf_(float x) { return 1.0f / (1.0f + __expf(-x)); }
__device__ __forceinline__ float siluf_(float x)    { return x * sigmoidf_(x); }

// ---------------- elementwise / prep kernels ----------------

__global__ void f32_to_f16_kernel(const float* __restrict__ src, _Float16* __restrict__ dst, int n) {
    int i = blockIdx.x * blockDim.x + threadIdx.x;
    if (i < n) dst[i] = (_Float16)src[i];
}

__global__ void embed_kernel(const int* __restrict__ ids, const float* __restrict__ mask,
                             const float* __restrict__ tok, const float* __restrict__ pos,
                             float* __restrict__ h) {
    int i = blockIdx.x * blockDim.x + threadIdx.x;
    if (i >= BLrows * DMODEL) return;
    int d = i % DMODEL;
    int row = i / DMODEL;          // b*L + l
    int l = row % L_;
    int id = ids[row];
    h[i] = (tok[(size_t)id * DMODEL + d] + pos[l * DMODEL + d]) * mask[row];
}

// f32 -> f16 row conversion with optional sequence flip (per batch)
__global__ void cvt_rows_kernel(const float* __restrict__ src, _Float16* __restrict__ dst,
                                int rowlen, int flip) {
    int i = blockIdx.x * blockDim.x + threadIdx.x;
    int n = BLrows * rowlen;
    if (i >= n) return;
    int c = i % rowlen;
    int l = (i / rowlen) % L_;
    int b = i / (rowlen * L_);
    int ls = flip ? (L_ - 1 - l) : l;
    dst[i] = (_Float16)src[((size_t)(b * L_ + ls)) * rowlen + c];
}

__global__ void copy_rows_f32_kernel(const float* __restrict__ src, float* __restrict__ dst,
                                     int rowlen, int flip) {
    int i = blockIdx.x * blockDim.x + threadIdx.x;
    int n = BLrows * rowlen;
    if (i >= n) return;
    int c = i % rowlen;
    int l = (i / rowlen) % L_;
    int b = i / (rowlen * L_);
    int ls = flip ? (L_ - 1 - l) : l;
    dst[i] = src[((size_t)(b * L_ + ls)) * rowlen + c];
}

// ---------------- WMMA GEMM: C[M,N] = A[M,K] * W[N,K]^T  (both K-contiguous) ----------------
// Block (8 waves) computes a 128x64 macro-tile. The 64 W rows used by the block are staged
// once in LDS via the Tensor Data Mover (TENSORcnt + s_wait_tensorcnt); OOB rows (ragged N)
// are zero-filled by TDM. Each wave: one 16-row A strip, 4 accumulators (16x64 of C), so one
// A fragment feeds 4 v_wmma issues. Requirements: M % 128 == 0, K % 32 == 0, K <= KMAX.
__global__ __launch_bounds__(256) void wmma_gemm_nt_kernel(
        const _Float16* __restrict__ A, const _Float16* __restrict__ W,
        float* __restrict__ C, int M, int N, int K) {
    __shared__ _Float16 sW[64 * KMAX];              // 64 KB

    const int lane = threadIdx.x & 31;
    const int wave = threadIdx.x >> 5;
    const int lm = lane & 15;
    const int hi = lane >> 4;

    const int tilesN64 = (N + 63) >> 6;
    const int bm = blockIdx.x / tilesN64;
    const int bn = blockIdx.x - bm * tilesN64;
    const int n0base = bn << 6;
    const int m0 = (bm << 7) + (wave << 4);

    // ---- stage rows [n0base, n0base+64) of W (each K halves) into LDS ----
#if HAS_TDM
    if (wave == 0) {
        unsigned lds_off = (unsigned)(size_t)(void*)&sW[0];               // LDS byte offset
        unsigned long long ga =
            (unsigned long long)(size_t)(const void*)(W + (size_t)n0base * K);
        unsigned dim1 = (unsigned)(N - n0base);                           // remaining rows (OOB -> 0)
        // D# group 0: count=1 | lds_addr | global_addr[31:0] | global_addr[56:32] + type=2
        u32x4 g0 = { 1u,
                     lds_off,
                     (unsigned)(ga & 0xffffffffull),
                     (unsigned)((ga >> 32) & 0x01ffffffull) | (2u << 30) };
        // D# group 1: data_size=2B; tensor_dim0=K; tensor_dim1=dim1; tile_dim0=K;
        //             tile_dim1=64; tile_dim2=0; tensor_dim0_stride=K
        i32x8 g1 = { (int)0x00010000u,
                     (int)(((unsigned)K & 0xffffu) << 16),
                     (int)((((unsigned)K >> 16) & 0xffffu) | ((dim1 & 0xffffu) << 16)),
                     (int)(((dim1 >> 16) & 0xffffu) | (((unsigned)K & 0xffffu) << 16)),
                     (int)64,
                     (int)(unsigned)K,
                     0, 0 };
        i32x4 gz = { 0, 0, 0, 0 };
        TDM_LOAD(g0, g1, gz, gz);
        WAIT_TENSORCNT0();
    }
#else
    // cooperative fallback: 8-half chunks, zero-fill ragged rows
    {
        const int chunks = K >> 3;
        for (int idx = threadIdx.x; idx < 64 * chunks; idx += 256) {
            int r = idx / chunks;
            int kc = (idx - r * chunks) << 3;
            v8h v = {};
            if (n0base + r < N) v = *(const v8h*)(W + (size_t)(n0base + r) * K + kc);
            *(v8h*)(&sW[r * K + kc]) = v;
        }
    }
#endif
    __syncthreads();

    const _Float16* arow = A + (size_t)(m0 + lm) * K;
    v8f acc0 = {}, acc1 = {}, acc2 = {}, acc3 = {};
    for (int k0 = 0; k0 < K; k0 += 32) {
        // A fragment: lane = row, element e -> k = k0 + hi*16 + e (matches B)
        v16h af = *(const v16h*)(arow + k0 + hi * 16);
        // B fragments from LDS: lane = output column, element e -> k = k0 + hi*16 + e
        const _Float16* brow = &sW[lm * K + k0 + hi * 16];
        v16h b0 = *(const v16h*)(brow);
        v16h b1 = *(const v16h*)(brow + 16 * K);
        v16h b2 = *(const v16h*)(brow + 32 * K);
        v16h b3 = *(const v16h*)(brow + 48 * K);
        acc0 = __builtin_amdgcn_wmma_f32_16x16x32_f16(false, af, false, b0, (short)0, acc0, false, false);
        acc1 = __builtin_amdgcn_wmma_f32_16x16x32_f16(false, af, false, b1, (short)0, acc1, false, false);
        acc2 = __builtin_amdgcn_wmma_f32_16x16x32_f16(false, af, false, b2, (short)0, acc2, false, false);
        acc3 = __builtin_amdgcn_wmma_f32_16x16x32_f16(false, af, false, b3, (short)0, acc3, false, false);
    }

    // C store (C/D layout: lane = column, VGPR r -> row hi*8 + r); bounds-check ragged N
#pragma unroll
    for (int t = 0; t < 4; ++t) {
        const v8f acc = (t == 0) ? acc0 : (t == 1) ? acc1 : (t == 2) ? acc2 : acc3;
        int n = n0base + t * 16 + lm;
        if (n < N) {
#pragma unroll
            for (int r = 0; r < 8; ++r)
                C[(size_t)(m0 + hi * 8 + r) * N + n] = acc[r];
        }
    }
}

// ---------------- dt = softplus(raw + dt_bias) ----------------
__global__ void dt_kernel(const float* __restrict__ zx, const float* __restrict__ dtb,
                          float* __restrict__ dt) {
    int i = blockIdx.x * blockDim.x + threadIdx.x;
    if (i >= BLrows * NHEADS) return;
    int hd = i % NHEADS;
    int row = i / NHEADS;
    float x = zx[(size_t)row * DINPROJ + DINNER + CONVDIM + hd] + dtb[hd];
    dt[i] = (x > 20.0f) ? x : log1pf(__expf(x));
}

// ---------------- depthwise causal conv (window 4) + bias + SiLU ----------------
__global__ void conv_silu_kernel(const float* __restrict__ zx, const float* __restrict__ cw,
                                 const float* __restrict__ cb, float* __restrict__ xbc) {
    int i = blockIdx.x * blockDim.x + threadIdx.x;
    if (i >= BLrows * CONVDIM) return;
    int c = i % CONVDIM;
    int l = (i / CONVDIM) % L_;
    int b = i / (CONVDIM * L_);
    float acc = cb[c];
#pragma unroll
    for (int k = 0; k < DCONV; ++k) {
        int ll = l - (DCONV - 1) + k;
        if (ll >= 0)
            acc += cw[c * DCONV + k] * zx[((size_t)(b * L_ + ll)) * DINPROJ + DINNER + c];
    }
    xbc[i] = siluf_(acc);
}

// ---------------- sequential SSM scan: one block per (batch, head) ----------------
__global__ __launch_bounds__(256) void mamba_scan_kernel(
        const float* __restrict__ xbc, const float* __restrict__ dt,
        const float* __restrict__ alog, const float* __restrict__ Dp,
        float* __restrict__ y) {
    int b = blockIdx.x / NHEADS;
    int hd = blockIdx.x % NHEADS;
    int t = threadIdx.x;
    int p = t >> 2;
    int sub = t & 3;
    float Aa = -__expf(alog[hd]);
    float Dv = Dp[hd];
    float hstate[16];
#pragma unroll
    for (int j = 0; j < 16; ++j) hstate[j] = 0.0f;
    __shared__ float sx[HEADDIM];
    __shared__ float sB[DSTATE];
    __shared__ float sC[DSTATE];
    __shared__ float sdt;
    for (int l = 0; l < L_; ++l) {
        size_t base = ((size_t)(b * L_ + l)) * CONVDIM;
        if (t < 64)                sx[t]        = xbc[base + hd * HEADDIM + t];
        else if (t < 128)          sB[t - 64]   = xbc[base + DINNER + (t - 64)];
        else if (t < 192)          sC[t - 128]  = xbc[base + DINNER + DSTATE + (t - 128)];
        else if (t == 192)         sdt          = dt[(size_t)(b * L_ + l) * NHEADS + hd];
        __syncthreads();
        float dtv = sdt;
        float dA  = __expf(dtv * Aa);
        float xv  = sx[p];
        float acc = 0.0f;
#pragma unroll
        for (int j = 0; j < 16; ++j) {
            int n = sub * 16 + j;
            hstate[j] = hstate[j] * dA + dtv * xv * sB[n];
            acc += hstate[j] * sC[n];
        }
        acc += __shfl_xor(acc, 1, 32);
        acc += __shfl_xor(acc, 2, 32);
        if (sub == 0)
            y[(size_t)(b * L_ + l) * DINNER + hd * HEADDIM + p] = acc + Dv * xv;
        __syncthreads();
    }
}

// ---------------- y * silu(z), RMSNorm over 512, emit f16 for out_proj ----------------
__global__ __launch_bounds__(256) void gate_rmsnorm_kernel(
        const float* __restrict__ yraw, const float* __restrict__ zx,
        const float* __restrict__ nw, _Float16* __restrict__ yh) {
    int row = blockIdx.x;
    int t = threadIdx.x;
    __shared__ float red[256];
    float z0 = zx[(size_t)row * DINPROJ + t];
    float z1 = zx[(size_t)row * DINPROJ + t + 256];
    float v0 = yraw[(size_t)row * DINNER + t]       * siluf_(z0);
    float v1 = yraw[(size_t)row * DINNER + t + 256] * siluf_(z1);
    red[t] = v0 * v0 + v1 * v1;
    __syncthreads();
    for (int s = 128; s > 0; s >>= 1) {
        if (t < s) red[t] += red[t + s];
        __syncthreads();
    }
    float scale = rsqrtf(red[0] / (float)DINNER + EPS_);
    yh[(size_t)row * DINNER + t]       = (_Float16)(v0 * scale * nw[t]);
    yh[(size_t)row * DINNER + t + 256] = (_Float16)(v1 * scale * nw[t + 256]);
}

// ---------------- concat(fwd, bwd) -> f16 for fusion GEMM ----------------
__global__ void combine_kernel(const float* __restrict__ fwd, const float* __restrict__ bwd,
                               _Float16* __restrict__ dst) {
    int i = blockIdx.x * blockDim.x + threadIdx.x;
    if (i >= BLrows * 2 * DMODEL) return;
    int c = i % (2 * DMODEL);
    int row = i / (2 * DMODEL);
    float v = (c < DMODEL) ? fwd[(size_t)row * DMODEL + c]
                           : bwd[(size_t)row * DMODEL + (c - DMODEL)];
    dst[i] = (_Float16)v;
}

// ---------------- sigmoid gate, fuse, LayerNorm -> output ----------------
__global__ __launch_bounds__(256) void fusion_final_kernel(
        const float* __restrict__ gp, const float* __restrict__ fb,
        const float* __restrict__ fwd, const float* __restrict__ bwd,
        const float* __restrict__ lnw, const float* __restrict__ lnb,
        float* __restrict__ out) {
    int row = blockIdx.x;
    int t = threadIdx.x;     // 256 threads == DMODEL
    __shared__ float red[256];
    float gf = sigmoidf_(gp[(size_t)row * 2 * DMODEL + t] + fb[t]);
    float gb = sigmoidf_(gp[(size_t)row * 2 * DMODEL + DMODEL + t] + fb[DMODEL + t]);
    float fused = gf * fwd[(size_t)row * DMODEL + t] + gb * bwd[(size_t)row * DMODEL + t];
    red[t] = fused;
    __syncthreads();
    for (int s = 128; s > 0; s >>= 1) {
        if (t < s) red[t] += red[t + s];
        __syncthreads();
    }
    float mu = red[0] / (float)DMODEL;
    __syncthreads();
    float d = fused - mu;
    red[t] = d * d;
    __syncthreads();
    for (int s = 128; s > 0; s >>= 1) {
        if (t < s) red[t] += red[t + s];
        __syncthreads();
    }
    float var = red[0] / (float)DMODEL;
    out[(size_t)row * DMODEL + t] = d * rsqrtf(var + EPS_) * lnw[t] + lnb[t];
}

// ---------------- host orchestration ----------------

static inline void launch_gemm(const _Float16* A, const _Float16* W, float* C,
                               int M, int N, int K, hipStream_t stream) {
    int tilesN64 = (N + 63) / 64;
    int blocks = (M / 128) * tilesN64;
    wmma_gemm_nt_kernel<<<blocks, 256, 0, stream>>>(A, W, C, M, N, K);
}

extern "C" void kernel_launch(void* const* d_in, const int* in_sizes, int n_in,
                              void* d_out, int out_size, void* d_ws, size_t ws_size,
                              hipStream_t stream) {
    (void)in_sizes; (void)n_in; (void)out_size; (void)ws_size;
    const int*   ids  = (const int*)d_in[0];
    const float* mask = (const float*)d_in[1];
    const float* tok  = (const float*)d_in[2];
    const float* pos  = (const float*)d_in[3];
    const float* inw  = (const float*)d_in[4];
    const float* cw   = (const float*)d_in[5];
    const float* cb   = (const float*)d_in[6];
    const float* dtb  = (const float*)d_in[7];
    const float* alog = (const float*)d_in[8];
    const float* Dp   = (const float*)d_in[9];
    const float* nw   = (const float*)d_in[10];
    const float* ow   = (const float*)d_in[11];
    const float* fw   = (const float*)d_in[12];
    const float* fb   = (const float*)d_in[13];
    const float* lnw  = (const float*)d_in[14];
    const float* lnb  = (const float*)d_in[15];
    float* out = (float*)d_out;

    // workspace carve-out (256B aligned blocks)
    char* p = (char*)d_ws;
    auto alloc = [&](size_t bytes) -> void* {
        void* r = (void*)p;
        p += (bytes + 255) & ~(size_t)255;
        return r;
    };
    float*    emb  = (float*)alloc((size_t)BLrows * DMODEL * 4);
    float*    u    = (float*)alloc((size_t)BLrows * DMODEL * 4);
    _Float16* uh   = (_Float16*)alloc((size_t)BLrows * DMODEL * 2);
    float*    zx   = (float*)alloc((size_t)BLrows * DINPROJ * 4);
    float*    xbc  = (float*)alloc((size_t)BLrows * CONVDIM * 4);
    float*    dts  = (float*)alloc((size_t)BLrows * NHEADS * 4);
    float*    yraw = (float*)alloc((size_t)BLrows * DINNER * 4);
    _Float16* yh   = (_Float16*)alloc((size_t)BLrows * DINNER * 2);
    float*    fwdb = (float*)alloc((size_t)BLrows * DMODEL * 4);
    float*    bwdb = (float*)alloc((size_t)BLrows * DMODEL * 4);
    _Float16* comb = (_Float16*)alloc((size_t)BLrows * 2 * DMODEL * 2);
    float*    gp   = (float*)alloc((size_t)BLrows * 2 * DMODEL * 4);
    _Float16* inwh = (_Float16*)alloc((size_t)4 * DINPROJ * DMODEL * 2);
    _Float16* owh  = (_Float16*)alloc((size_t)4 * DMODEL * DINNER * 2);
    _Float16* fwh  = (_Float16*)alloc((size_t)2 * DMODEL * 2 * DMODEL * 2);

    const int TB = 256;
    auto grid1d = [](int n) { return (n + 255) / 256; };

    // convert weights to f16 once per launch (deterministic)
    f32_to_f16_kernel<<<grid1d(4 * DINPROJ * DMODEL), TB, 0, stream>>>(inw, inwh, 4 * DINPROJ * DMODEL);
    f32_to_f16_kernel<<<grid1d(4 * DMODEL * DINNER), TB, 0, stream>>>(ow, owh, 4 * DMODEL * DINNER);
    f32_to_f16_kernel<<<grid1d(2 * DMODEL * 2 * DMODEL), TB, 0, stream>>>(fw, fwh, 2 * DMODEL * 2 * DMODEL);

    // embedding (+mask)
    embed_kernel<<<grid1d(BLrows * DMODEL), TB, 0, stream>>>(ids, mask, tok, pos, emb);

    for (int d = 0; d < 2; ++d) {
        for (int layer = 0; layer < 2; ++layer) {
            const int wi = d * 2 + layer;
            const float* src = (layer == 0) ? emb : u;
            const int flip = (layer == 0 && d == 1) ? 1 : 0;
            // f16 (and optionally flipped) layer input
            cvt_rows_kernel<<<grid1d(BLrows * DMODEL), TB, 0, stream>>>(src, uh, DMODEL, flip);
            // in_proj: (2048 x 1160) = uh (2048 x 256) * in_w^T
            launch_gemm(uh, inwh + (size_t)wi * DINPROJ * DMODEL, zx,
                        BLrows, DINPROJ, DMODEL, stream);
            // dt = softplus(raw + bias)
            dt_kernel<<<grid1d(BLrows * NHEADS), TB, 0, stream>>>(zx, dtb + wi * NHEADS, dts);
            // depthwise conv + bias + SiLU
            conv_silu_kernel<<<grid1d(BLrows * CONVDIM), TB, 0, stream>>>(
                zx, cw + (size_t)wi * CONVDIM * DCONV, cb + (size_t)wi * CONVDIM, xbc);
            // sequential SSM scan
            mamba_scan_kernel<<<B_ * NHEADS, 256, 0, stream>>>(
                xbc, dts, alog + wi * NHEADS, Dp + wi * NHEADS, yraw);
            // SiLU(z) gating + RMSNorm -> f16
            gate_rmsnorm_kernel<<<BLrows, 256, 0, stream>>>(yraw, zx, nw + (size_t)wi * DINNER, yh);
            // out_proj: (2048 x 256) = yh (2048 x 512) * ow^T
            launch_gemm(yh, owh + (size_t)wi * DMODEL * DINNER, u,
                        BLrows, DMODEL, DINNER, stream);
        }
        if (d == 0)
            copy_rows_f32_kernel<<<grid1d(BLrows * DMODEL), TB, 0, stream>>>(u, fwdb, DMODEL, 0);
        else
            copy_rows_f32_kernel<<<grid1d(BLrows * DMODEL), TB, 0, stream>>>(u, bwdb, DMODEL, 1);
    }

    // fusion gate
    combine_kernel<<<grid1d(BLrows * 2 * DMODEL), TB, 0, stream>>>(fwdb, bwdb, comb);
    launch_gemm(comb, fwh, gp, BLrows, 2 * DMODEL, 2 * DMODEL, stream);
    fusion_final_kernel<<<BLrows, 256, 0, stream>>>(gp, fb, fwdb, bwdb, lnw, lnb, out);
}
